// PLSTM_73186242724133
// MI455X (gfx1250) — matrix operand order repaired and checked by reading, not compiled, not run on hardware
//
#include <hip/hip_runtime.h>
#include <hip/hip_bf16.h>
#include <math.h>

typedef __bf16 bf16_t;
typedef __bf16 v16bf __attribute__((ext_vector_type(16)));
typedef __bf16 v8bf  __attribute__((ext_vector_type(8)));
typedef float  v8f   __attribute__((ext_vector_type(8)));
typedef unsigned int v4u __attribute__((ext_vector_type(4)));
typedef int v4i __attribute__((ext_vector_type(4)));
typedef int v8i __attribute__((ext_vector_type(8)));

#define T_STEPS   4096
#define IN_DIM    1024
#define MEM_DIM   1024
#define GATE_DIM  4096            // 4 * MEM_DIM
#define NB        32              // persistent blocks in scan
#define CELLS_PB  32              // MEM_DIM / NB
#define ROWS_PB   128             // 4 gates * CELLS_PB

// TDM-padded LDS row: 1024 bf16 + 16B pad after each 512 elements -> 2080 B
#define WH_ROW_BYTES 2080
#define WH_BYTES    (ROWS_PB * WH_ROW_BYTES)          // 266240
#define HS_OFF      (WH_BYTES)                        // bf16[1024] -> 2048 B
#define GATES_OFF   (HS_OFF + 2048)                   // float[128] -> 512 B
#define CSTATE_OFF  (GATES_OFF + 512)                 // float[32]  -> 128 B
#define SMEM_BYTES  (CSTATE_OFF + 128)                // 268928 B (< 320 KB)

// ---- workspace layout ----
#define XPROJ_BYTES ((size_t)T_STEPS * GATE_DIM * 4)  // 64 MB
#define CVT_BYTES   ((size_t)4194304 * 2)             // 8 MB per bf16 array
#define INB_OFF     (XPROJ_BYTES)
#define WXB_OFF     (XPROJ_BYTES + CVT_BYTES)
#define WHB_OFF     (XPROJ_BYTES + 2 * CVT_BYTES)
#define HBF_OFF     (XPROJ_BYTES + 3 * CVT_BYTES)
#define CTR_OFF     (HBF_OFF + 8192)

__device__ __forceinline__ float sigmoidf_(float x) {
  return 1.0f / (1.0f + __expf(-x));
}

// -------- fp32 -> bf16 conversion of inputs, Wx, Wh --------
__global__ void plstm_cvt(const float* __restrict__ inp, const float* __restrict__ wx,
                          const float* __restrict__ wh,
                          bf16_t* __restrict__ inb, bf16_t* __restrict__ wxb,
                          bf16_t* __restrict__ whb) {
  int i = blockIdx.x * blockDim.x + threadIdx.x;   // 0 .. 4194303
  inb[i] = (bf16_t)inp[i];
  wxb[i] = (bf16_t)wx[i];
  whb[i] = (bf16_t)wh[i];
}

// -------- zero the grid-barrier counter (fresh every launch) --------
__global__ void plstm_init(unsigned* __restrict__ ctr) {
  if (threadIdx.x == 0 && blockIdx.x == 0) *ctr = 0u;
}

// -------- x_proj GEMM: (T,IN) x (GATE,IN)^T + bx + bh, bf16 WMMA, f32 acc --------
// One wave computes a 16(M) x 64(N) strip; register double-buffer overlaps
// global loads with WMMA so the compiler need not wait loadcnt==0 per wmma.
__global__ void plstm_gemm(const bf16_t* __restrict__ A,   // inputs bf16 (T, IN)
                           const bf16_t* __restrict__ B,   // Wx bf16 (GATE, IN)
                           const float* __restrict__ bx, const float* __restrict__ bh,
                           float* __restrict__ xproj) {
  const int tid  = threadIdx.x;
  const int wv   = tid >> 5;
  const int lane = tid & 31;
  const int half = lane >> 4;
  const int l15  = lane & 15;

  const int job = blockIdx.x * 8 + wv;       // 16384 jobs: 256 M-tiles x 64 N-groups
  const int tM  = (job >> 6) << 4;
  const int tN0 = (job & 63) << 6;

  const bf16_t* arow  = A + (size_t)(tM + l15) * IN_DIM + half * 16;
  const bf16_t* brow0 = B + (size_t)(tN0 +  0 + l15) * IN_DIM + half * 16;
  const bf16_t* brow1 = B + (size_t)(tN0 + 16 + l15) * IN_DIM + half * 16;
  const bf16_t* brow2 = B + (size_t)(tN0 + 32 + l15) * IN_DIM + half * 16;
  const bf16_t* brow3 = B + (size_t)(tN0 + 48 + l15) * IN_DIM + half * 16;

  v8f acc0 = {}, acc1 = {}, acc2 = {}, acc3 = {};

  v8bf alo = *(const v8bf*)(arow);
  v8bf ahi = *(const v8bf*)(arow + 8);
  v16bf b0 = *(const v16bf*)(brow0);
  v16bf b1 = *(const v16bf*)(brow1);
  v16bf b2 = *(const v16bf*)(brow2);
  v16bf b3 = *(const v16bf*)(brow3);

  for (int k0 = 0; k0 < IN_DIM - 32; k0 += 32) {
    const int nk = k0 + 32;
    // prefetch next K-slice fragments (keeps loads in flight across the WMMAs)
    v8bf nalo = *(const v8bf*)(arow + nk);
    v8bf nahi = *(const v8bf*)(arow + nk + 8);
    v16bf nb0 = *(const v16bf*)(brow0 + nk);
    v16bf nb1 = *(const v16bf*)(brow1 + nk);
    v16bf nb2 = *(const v16bf*)(brow2 + nk);
    v16bf nb3 = *(const v16bf*)(brow3 + nk);

    v16bf a;
#pragma unroll
    for (int i = 0; i < 8; ++i) { a[i] = alo[i]; a[i + 8] = ahi[i]; }
    acc0 = __builtin_amdgcn_wmma_f32_16x16x32_bf16(false, a, false, b0, (short)0, acc0, false, false);
    acc1 = __builtin_amdgcn_wmma_f32_16x16x32_bf16(false, a, false, b1, (short)0, acc1, false, false);
    acc2 = __builtin_amdgcn_wmma_f32_16x16x32_bf16(false, a, false, b2, (short)0, acc2, false, false);
    acc3 = __builtin_amdgcn_wmma_f32_16x16x32_bf16(false, a, false, b3, (short)0, acc3, false, false);

    alo = nalo; ahi = nahi; b0 = nb0; b1 = nb1; b2 = nb2; b3 = nb3;
  }
  {
    v16bf a;
#pragma unroll
    for (int i = 0; i < 8; ++i) { a[i] = alo[i]; a[i + 8] = ahi[i]; }
    acc0 = __builtin_amdgcn_wmma_f32_16x16x32_bf16(false, a, false, b0, (short)0, acc0, false, false);
    acc1 = __builtin_amdgcn_wmma_f32_16x16x32_bf16(false, a, false, b1, (short)0, acc1, false, false);
    acc2 = __builtin_amdgcn_wmma_f32_16x16x32_bf16(false, a, false, b2, (short)0, acc2, false, false);
    acc3 = __builtin_amdgcn_wmma_f32_16x16x32_bf16(false, a, false, b3, (short)0, acc3, false, false);
  }

#pragma unroll
  for (int c = 0; c < 4; ++c) {
    const v8f acc = (c == 0) ? acc0 : (c == 1) ? acc1 : (c == 2) ? acc2 : acc3;
    const int n = tN0 + c * 16 + l15;
    const float bias = bx[n] + bh[n];
#pragma unroll
    for (int r = 0; r < 8; ++r) {
      const int m = r + half * 8;
      xproj[(size_t)(tM + m) * GATE_DIM + n] = acc[r] + bias;
    }
  }
}

// -------- TDM: load one 32-row x 1024-elem bf16 tile into LDS with padding --------
// LDS padding: 16 B (4 dwords, pad_amount enc=3) after every 256 dwords
// (pad_interval enc=7, i.e. 512 bf16) -> row stride 2080 B, 16B-aligned chunks.
__device__ __forceinline__ void tdm_load_tile(unsigned lds_addr, const bf16_t* gsrc) {
#if __has_builtin(__builtin_amdgcn_tensor_load_to_lds)
  const unsigned long long ga = (unsigned long long)(uintptr_t)gsrc;
  v4u g0;
  g0[0] = 1u;                                   // count=1, user mode
  g0[1] = lds_addr;                             // lds_addr[31:0]
  g0[2] = (unsigned)(ga & 0xFFFFFFFFull);       // global_addr[95:64]
  g0[3] = (unsigned)((ga >> 32) & 0x01FFFFFFull) | (2u << 30);  // addr hi | type=2
  v8i g1;
  g1[0] = (int)((1u << 16)        // data_size = 2 bytes
              | (1u << 20)        // pad_enable
              | (7u << 22)        // pad_interval: 256 dwords
              | (3u << 25));      // pad_amount: 4 dwords (16 B)
  g1[1] = (int)(1024u << 16);     // tensor_dim0 = 1024 (low 16 in [63:48])
  g1[2] = (int)(32u << 16);       // tensor_dim0 hi=0 | tensor_dim1 = 32
  g1[3] = (int)(1024u << 16);     // tensor_dim1 hi=0 | tile_dim0 = 1024
  g1[4] = 32;                     // tile_dim1 = 32 | tile_dim2 = 0
  g1[5] = 1024;                   // tensor_dim0_stride low 32
  g1[6] = 0;                      // stride0 hi | tensor_dim1_stride low
  g1[7] = 0;
  v4i g2 = {0, 0, 0, 0};
  v4i g3 = {0, 0, 0, 0};
#if __clang_major__ >= 23
  v8i g4 = {0, 0, 0, 0, 0, 0, 0, 0};
  __builtin_amdgcn_tensor_load_to_lds(g0, g1, g2, g3, g4, 0);
#else
  __builtin_amdgcn_tensor_load_to_lds(g0, g1, g2, g3, 0);
#endif
#else
  (void)lds_addr; (void)gsrc;
#endif
}

// -------- persistent recurrent scan: Wh resident in LDS, WMMA matvec per step --------
__global__ void plstm_scan(const bf16_t* __restrict__ whb, const float* __restrict__ xproj,
                           const float* __restrict__ zc, bf16_t* __restrict__ hbf,
                           unsigned* __restrict__ ctr, float* __restrict__ out) {
  extern __shared__ char smem[];
  bf16_t* h_s   = (bf16_t*)(smem + HS_OFF);
  float*  gates = (float*)(smem + GATES_OFF);
  float*  c_s   = (float*)(smem + CSTATE_OFF);

  const int b   = blockIdx.x;
  const int tid = threadIdx.x;
  const int wv   = tid >> 5;
  const int lane = tid & 31;
  const int half = lane >> 4;
  const int l15  = lane & 15;

  // Stage this block's 128 Wh rows (4 gates x 32 cells) into LDS via the
  // Tensor Data Mover: one 2D tile per gate group, TDM inserts the LDS padding.
#if __has_builtin(__builtin_amdgcn_tensor_load_to_lds)
  if (wv == 0) {
    const unsigned lds_base = (unsigned)(uintptr_t)(void*)smem;
#pragma unroll
    for (int q = 0; q < 4; ++q) {
      tdm_load_tile(lds_base + (unsigned)q * 32u * WH_ROW_BYTES,
                    whb + (size_t)(q * MEM_DIM + b * CELLS_PB) * MEM_DIM);
    }
    __builtin_amdgcn_s_wait_tensorcnt(0);
  }
#else
  for (int idx = tid; idx < ROWS_PB * IN_DIM; idx += 256) {
    const int lr = idx >> 10, k = idx & 1023;
    char* dst = smem + (size_t)lr * WH_ROW_BYTES + k * 2 + ((k >> 9) << 4);
    *(bf16_t*)dst = whb[(size_t)((lr >> 5) * MEM_DIM + b * CELLS_PB + (lr & 31)) * MEM_DIM + k];
  }
#endif
  for (int k = tid; k < MEM_DIM; k += 256) h_s[k] = (bf16_t)0.0f;
  if (tid < CELLS_PB) c_s[tid] = 0.0f;
  const float zval = (tid < CELLS_PB) ? zc[b * CELLS_PB + tid] : 0.0f;
  __syncthreads();

  const char* wrowb = smem + (size_t)(wv * 16 + l15) * WH_ROW_BYTES;

  for (int t = 0; t < T_STEPS; ++t) {
    // matvec: 16 rows per wave, h broadcast into every B column -> result
    // replicated across lanes, no cross-lane reduction needed.
    v8f acc = {};
    for (int k0 = 0; k0 < MEM_DIM; k0 += 32) {
      const int off = (k0 + half * 16) * 2 + ((k0 >> 9) << 4);  // TDM pad shift
      v8bf alo = *(const v8bf*)(wrowb + off);
      v8bf ahi = *(const v8bf*)(wrowb + off + 16);
      v16bf a;
#pragma unroll
      for (int i = 0; i < 8; ++i) { a[i] = alo[i]; a[i + 8] = ahi[i]; }
      v16bf hb = *(const v16bf*)(h_s + k0 + half * 16);
      acc = __builtin_amdgcn_wmma_f32_16x16x32_bf16(false, a, false, hb, (short)0, acc, false, false);
    }
    if (l15 == 0) {
      const float* xp = xproj + (size_t)t * GATE_DIM;
#pragma unroll
      for (int r = 0; r < 8; ++r) {
        const int lr = wv * 16 + r + half * 8;
        const int g  = (lr >> 5) * MEM_DIM + b * CELLS_PB + (lr & 31);
        gates[lr] = acc[r] + xp[g];
      }
      if (t + 1 < T_STEPS) {  // prefetch next timestep's gate row segment
        const int lr0 = wv * 16 + half * 8;
        const int g0  = (lr0 >> 5) * MEM_DIM + b * CELLS_PB + (lr0 & 31);
        __builtin_prefetch(xproj + (size_t)(t + 1) * GATE_DIM + g0, 0, 0);
      }
    }
    __syncthreads();

    if (tid < CELLS_PB) {
      const int m = tid;
      const float ig = sigmoidf_(gates[m]);
      const float og = sigmoidf_(gates[CELLS_PB + m]);       // 32 + m
      const float fg = sigmoidf_(gates[2 * CELLS_PB + m]);   // 64 + m
      const float cp = c_s[m];
      const float ug = tanhf(gates[3 * CELLS_PB + m] + zval * cp);  // 96 + m
      const float c  = ig * ug + fg * cp;
      const float h  = og * tanhf(c);
      c_s[m] = c;
      hbf[(size_t)(t & 1) * MEM_DIM + b * CELLS_PB + m] = (bf16_t)h;
      if (t == T_STEPS - 1) {
        out[b * CELLS_PB + m] = c;                 // c_T
        out[MEM_DIM + b * CELLS_PB + m] = h;       // h_T
      }
    }
    __syncthreads();

    // one grid barrier per step (double-buffered h makes a single barrier safe)
    if (tid == 0) {
      __threadfence();
      atomicAdd(ctr, 1u);
      const unsigned target = (unsigned)(t + 1) * NB;
      while (__hip_atomic_load(ctr, __ATOMIC_ACQUIRE, __HIP_MEMORY_SCOPE_AGENT) < target)
        __builtin_amdgcn_s_sleep(1);
    }
    __syncthreads();

    if (t < T_STEPS - 1) {
      volatile const bf16_t* src = hbf + (size_t)(t & 1) * MEM_DIM;
      for (int k = tid; k < MEM_DIM; k += 256) h_s[k] = src[k];
      __syncthreads();
    }
  }
}

extern "C" void kernel_launch(void* const* d_in, const int* in_sizes, int n_in,
                              void* d_out, int out_size, void* d_ws, size_t ws_size,
                              hipStream_t stream) {
  (void)in_sizes; (void)n_in; (void)out_size; (void)ws_size;
  const float* inputs = (const float*)d_in[0];
  const float* Wx     = (const float*)d_in[1];
  const float* bx     = (const float*)d_in[2];
  const float* Wh     = (const float*)d_in[3];
  const float* bh     = (const float*)d_in[4];
  const float* zc     = (const float*)d_in[5];
  float* out = (float*)d_out;

  char* ws = (char*)d_ws;
  float*   xproj = (float*)ws;
  bf16_t*  inb   = (bf16_t*)(ws + INB_OFF);
  bf16_t*  wxb   = (bf16_t*)(ws + WXB_OFF);
  bf16_t*  whb   = (bf16_t*)(ws + WHB_OFF);
  bf16_t*  hbf   = (bf16_t*)(ws + HBF_OFF);
  unsigned* ctr  = (unsigned*)(ws + CTR_OFF);

  (void)hipFuncSetAttribute((const void*)plstm_scan,
                            hipFuncAttributeMaxDynamicSharedMemorySize, (int)SMEM_BYTES);

  plstm_cvt<<<(T_STEPS * IN_DIM) / 256, 256, 0, stream>>>(inputs, Wx, Wh, inb, wxb, whb);
  plstm_init<<<1, 64, 0, stream>>>(ctr);
  plstm_gemm<<<2048, 256, 0, stream>>>(inb, wxb, bx, bh, xproj);
  plstm_scan<<<NB, 256, SMEM_BYTES, stream>>>(whb, xproj, zc, hbf, ctr, out);
}
